// SuperpointNeuralOperator_23613730193453
// MI455X (gfx1250) — matrix operand order repaired and checked by reading, not compiled, not run on hardware
//
#include <hip/hip_runtime.h>
#include <hip/hip_bf16.h>
#include <math.h>

// Problem constants (match reference)
#define NPTS 100000
#define KNB  16
#define HD   64
#define TLAY 3
#define NSP  512
#define KANC 8
#define EPSLN 1e-5f

typedef _Float16 v16h __attribute__((ext_vector_type(16)));
typedef _Float16 v8h  __attribute__((ext_vector_type(8)));
typedef float    v8f  __attribute__((ext_vector_type(8)));

// ---------------------------------------------------------------------------
// Helpers
// ---------------------------------------------------------------------------
__device__ __forceinline__ void wave_lds_sync() {
  // Wave-local LDS ordering: LDS is in-order per wave at the HW level; the
  // explicit dscnt wait + compiler memory barrier guarantee cross-lane
  // visibility of prior ds_store before subsequent ds_load.
#if __has_builtin(__builtin_amdgcn_s_wait_dscnt)
  __builtin_amdgcn_s_wait_dscnt(0);
#else
  asm volatile("s_wait_dscnt 0" ::: "memory");
#endif
  asm volatile("" ::: "memory");
}

__device__ __forceinline__ float gelu_exact(float x) {
  return 0.5f * x * (1.0f + erff(x * 0.70710678118654752440f));
}

__device__ __forceinline__ v8f bcast8(float x) {
  v8f r;
#pragma unroll
  for (int i = 0; i < 8; ++i) r[i] = x;
  return r;
}

__device__ __forceinline__ v8f wmma_f16(v16h a, v16h b, v8f c) {
  return __builtin_amdgcn_wmma_f32_16x16x32_f16(false, a, false, b, (short)0, c,
                                                false, false);
}

// Load a 16x32 f16 A-fragment from an LDS-resident row-major matrix.
// ISA layout: lanes 0-15 -> row=lane, K = kc*32 + {0..7, 16..23}
//             lanes16-31 -> row=lane-16, K = kc*32 + {8..15, 24..31}
__device__ __forceinline__ v16h load_a_frag(const _Float16* base, int rowStride,
                                            int kc, int lane) {
  const int row = lane & 15;
  const int sel = (lane >> 4) << 3;  // 0 or 8
  const _Float16* p0 = base + row * rowStride + kc * 32 + sel;
  const v8h lo = *(const v8h*)p0;         // ds_load_b128
  const v8h hi = *(const v8h*)(p0 + 16);  // ds_load_b128
  v16h r;
#pragma unroll
  for (int j = 0; j < 8; ++j) { r[j] = lo[j]; r[8 + j] = hi[j]; }
  return r;
}

// ---------------------------------------------------------------------------
// Weight packing: W[K][64] f32 -> f16 B-fragments, frag index (kc*4+nt)*32+lane,
// 16 contiguous halves per lane. lane<16: K=kc*32+{0..7,16..23} ; lane>=16: K=kc*32+{8..15,24..31}.
// ---------------------------------------------------------------------------
__global__ void sno_pack_kernel(const float* __restrict__ W,
                                _Float16* __restrict__ dst, int ktrue) {
  const int kc = blockIdx.x;
  const int nt = threadIdx.x >> 5;
  const int lane = threadIdx.x & 31;
  const int col = nt * 16 + (lane & 15);
  const int kbase = kc * 32 + ((lane >= 16) ? 8 : 0);
  _Float16* o = dst + ((size_t)(kc * 4 + nt) * 32 + lane) * 16;
#pragma unroll
  for (int j = 0; j < 16; ++j) {
    const int kk = kbase + (j < 8 ? j : j + 8);
    o[j] = (kk < ktrue) ? (_Float16)W[kk * HD + col] : (_Float16)0.0f;
  }
}

// ---------------------------------------------------------------------------
// Lift: v = [coords|feat] @ lift_w + lift_b   (tiny; VALU)
// ---------------------------------------------------------------------------
__global__ void __launch_bounds__(256)
sno_lift_kernel(const float* __restrict__ coords, const float* __restrict__ feat,
                const float* __restrict__ lw, const float* __restrict__ lb,
                float* __restrict__ vout) {
  const int t = blockIdx.x * 256 + threadIdx.x;
  if (t >= NPTS * HD) return;
  const int n = t >> 6, h = t & 63;
  float s = lb[h];
  s += coords[n * 3 + 0] * lw[0 * HD + h];
  s += coords[n * 3 + 1] * lw[1 * HD + h];
  s += coords[n * 3 + 2] * lw[2 * HD + h];
  s += feat[n * 3 + 0] * lw[3 * HD + h];
  s += feat[n * 3 + 1] * lw[4 * HD + h];
  s += feat[n * 3 + 2] * lw[5 * HD + h];
  vout[t] = s;
}

// ---------------------------------------------------------------------------
// Green-function MLP + neighbor reduce. One wave = one point (16 neighbor rows).
// A = [rel_pos(3) | v_i(64) | v_j(64) | pad -> 160]; h1=gelu(A@g1+b1);
// h2=gelu(h1@g2+b2); G=sigmoid(h2@g3+b3); integral = mean_k G*v_j.
// ---------------------------------------------------------------------------
__global__ void __launch_bounds__(128)
sno_green_kernel(const float* __restrict__ v, const float* __restrict__ coords,
                 const int* __restrict__ idx, const _Float16* __restrict__ pk1,
                 const float* __restrict__ b1, const _Float16* __restrict__ pk2,
                 const float* __restrict__ b2, const float* __restrict__ w3,
                 const float* __restrict__ b3, float* __restrict__ integral) {
  __shared__ _Float16 shA[4][16 * 160];
  __shared__ float shVJ[4][16 * HD];
  __shared__ _Float16 shH[4][16 * HD];
  __shared__ float shG[4][16];

  const int lane = threadIdx.x & 31;
  const int w = threadIdx.x >> 5;
  const int p = blockIdx.x * 4 + w;
  if (p >= NPTS) return;

  _Float16* A = shA[w];
  float* vj = shVJ[w];
  _Float16* h1 = shH[w];
  float* G = shG[w];

  const float vp0 = v[p * HD + lane];
  const float vp1 = v[p * HD + lane + 32];

#pragma unroll 4
  for (int kk = 0; kk < KNB; ++kk) {
    const int j = idx[p * KNB + kk];
    if (lane < 3)
      A[kk * 160 + lane] = (_Float16)(coords[j * 3 + lane] - coords[p * 3 + lane]);
    A[kk * 160 + 3 + lane] = (_Float16)vp0;
    A[kk * 160 + 3 + lane + 32] = (_Float16)vp1;
    const float a0 = v[j * HD + lane];
    const float a1 = v[j * HD + lane + 32];
    A[kk * 160 + 67 + lane] = (_Float16)a0;
    A[kk * 160 + 67 + lane + 32] = (_Float16)a1;
    vj[kk * HD + lane] = a0;
    vj[kk * HD + lane + 32] = a1;
    if (lane < 29) A[kk * 160 + 131 + lane] = (_Float16)0.0f;  // K pad 131..159
  }
  wave_lds_sync();

  v16h af[5];
#pragma unroll
  for (int kc = 0; kc < 5; ++kc) af[kc] = load_a_frag(A, 160, kc, lane);

  const int colb = lane & 15;
  const int rb = (lane >> 4) << 3;

  // Layer 1: 16x160 @ 160x64  (20 WMMAs)
#pragma unroll
  for (int nt = 0; nt < 4; ++nt) {
    v8f c = bcast8(b1[nt * 16 + colb]);
#pragma unroll
    for (int kc = 0; kc < 5; ++kc) {
      const v16h bf = ((const v16h*)pk1)[(kc * 4 + nt) * 32 + lane];
      c = wmma_f16(af[kc], bf, c);
    }
    const int col = nt * 16 + colb;
#pragma unroll
    for (int r = 0; r < 8; ++r)
      h1[(rb + r) * HD + col] = (_Float16)gelu_exact(c[r]);
  }
  wave_lds_sync();

  // Layer 2: 16x64 @ 64x64  (8 WMMAs); h2 overwrites h1 staging after frags read
  const v16h ha0 = load_a_frag(h1, HD, 0, lane);
  const v16h ha1 = load_a_frag(h1, HD, 1, lane);
#pragma unroll
  for (int nt = 0; nt < 4; ++nt) {
    v8f c = bcast8(b2[nt * 16 + colb]);
    c = wmma_f16(ha0, ((const v16h*)pk2)[(0 * 4 + nt) * 32 + lane], c);
    c = wmma_f16(ha1, ((const v16h*)pk2)[(1 * 4 + nt) * 32 + lane], c);
    const int col = nt * 16 + colb;
#pragma unroll
    for (int r = 0; r < 8; ++r)
      h1[(rb + r) * HD + col] = (_Float16)gelu_exact(c[r]);
  }
  wave_lds_sync();

  // Head: G = sigmoid(h2 @ g3_w + b3)  (16 rows, VALU)
  if (lane < 16) {
    float s = b3[0];
#pragma unroll 8
    for (int jj = 0; jj < HD; ++jj) s += (float)h1[lane * HD + jj] * w3[jj];
    G[lane] = 1.0f / (1.0f + expf(-s));
  }
  wave_lds_sync();

  // integral = mean_k G[k] * v_j[k]  (f32, 2 cols/lane)
  float s0 = 0.f, s1 = 0.f;
#pragma unroll
  for (int kk = 0; kk < KNB; ++kk) {
    const float g = G[kk];
    s0 += g * vj[kk * HD + lane];
    s1 += g * vj[kk * HD + lane + 32];
  }
  integral[p * HD + lane] = s0 * (1.0f / KNB);
  integral[p * HD + lane + 32] = s1 * (1.0f / KNB);
}

// ---------------------------------------------------------------------------
// v = LayerNorm(relu(integral + v @ W_w)); one wave = 16 rows (WMMA GEMM).
// ---------------------------------------------------------------------------
__global__ void __launch_bounds__(64)
sno_update_kernel(float* __restrict__ v, const float* __restrict__ integral,
                  const _Float16* __restrict__ pkw, const float* __restrict__ lng,
                  const float* __restrict__ lnb) {
  __shared__ _Float16 shAv[2][16 * HD];
  __shared__ float shU[2][16 * HD];
  __shared__ float shMu[2][16];
  __shared__ float shRi[2][16];

  const int lane = threadIdx.x & 31;
  const int w = threadIdx.x >> 5;
  const int p0 = (blockIdx.x * 2 + w) * 16;
  if (p0 >= NPTS) return;

  _Float16* As = shAv[w];
  float* U = shU[w];

#pragma unroll 4
  for (int r = 0; r < 16; ++r) {
    As[r * HD + lane] = (_Float16)v[(p0 + r) * HD + lane];
    As[r * HD + lane + 32] = (_Float16)v[(p0 + r) * HD + lane + 32];
  }
  wave_lds_sync();

  const v16h a0 = load_a_frag(As, HD, 0, lane);
  const v16h a1 = load_a_frag(As, HD, 1, lane);
  const int colb = lane & 15;
  const int rb = (lane >> 4) << 3;

#pragma unroll
  for (int nt = 0; nt < 4; ++nt) {
    v8f c = bcast8(0.0f);
    c = wmma_f16(a0, ((const v16h*)pkw)[(0 * 4 + nt) * 32 + lane], c);
    c = wmma_f16(a1, ((const v16h*)pkw)[(1 * 4 + nt) * 32 + lane], c);
    const int col = nt * 16 + colb;
#pragma unroll
    for (int r = 0; r < 8; ++r) {
      const int row = rb + r;
      const float u = c[r] + integral[(p0 + row) * HD + col];
      U[row * HD + col] = fmaxf(u, 0.0f);
    }
  }
  wave_lds_sync();

  if (lane < 16) {
    float mu = 0.f;
#pragma unroll 8
    for (int cc = 0; cc < HD; ++cc) mu += U[lane * HD + cc];
    mu *= (1.0f / HD);
    float var = 0.f;
#pragma unroll 8
    for (int cc = 0; cc < HD; ++cc) {
      const float d = U[lane * HD + cc] - mu;
      var += d * d;
    }
    var *= (1.0f / HD);
    shMu[w][lane] = mu;
    shRi[w][lane] = rsqrtf(var + EPSLN);
  }
  wave_lds_sync();

#pragma unroll 4
  for (int r = 0; r < 16; ++r) {
    const float mu = shMu[w][r], ri = shRi[w][r];
    const int c0 = lane, c1 = lane + 32;
    v[(p0 + r) * HD + c0] = (U[r * HD + c0] - mu) * ri * lng[c0] + lnb[c0];
    v[(p0 + r) * HD + c1] = (U[r * HD + c1] - mu) * ri * lng[c1] + lnb[c1];
  }
}

// ---------------------------------------------------------------------------
// Generic 16-row WMMA GEMM: out = gather(src) @ Wpacked + bias  (keys/queries)
// ---------------------------------------------------------------------------
__global__ void __launch_bounds__(64)
sno_gemm16_kernel(const float* __restrict__ src, const _Float16* __restrict__ pk,
                  const float* __restrict__ bias, float* __restrict__ out,
                  const int* __restrict__ gidx, int nrows) {
  __shared__ _Float16 shAv[2][16 * HD];
  const int lane = threadIdx.x & 31;
  const int w = threadIdx.x >> 5;
  const int p0 = (blockIdx.x * 2 + w) * 16;
  if (p0 >= nrows) return;
  _Float16* As = shAv[w];

#pragma unroll 4
  for (int r = 0; r < 16; ++r) {
    int row = p0 + r;
    if (row >= nrows) row = nrows - 1;
    const int g = gidx ? gidx[row] : row;
    As[r * HD + lane] = (_Float16)src[g * HD + lane];
    As[r * HD + lane + 32] = (_Float16)src[g * HD + lane + 32];
  }
  wave_lds_sync();

  const v16h a0 = load_a_frag(As, HD, 0, lane);
  const v16h a1 = load_a_frag(As, HD, 1, lane);
  const int colb = lane & 15;
  const int rb = (lane >> 4) << 3;

#pragma unroll
  for (int nt = 0; nt < 4; ++nt) {
    v8f c = bcast8(bias[nt * 16 + colb]);
    c = wmma_f16(a0, ((const v16h*)pk)[(0 * 4 + nt) * 32 + lane], c);
    c = wmma_f16(a1, ((const v16h*)pk)[(1 * 4 + nt) * 32 + lane], c);
    const int col = nt * 16 + colb;
#pragma unroll
    for (int r = 0; r < 8; ++r) {
      const int row = p0 + rb + r;
      if (row < nrows) out[row * HD + col] = c[r];
    }
  }
}

// ---------------------------------------------------------------------------
// scores = ||v||
// ---------------------------------------------------------------------------
__global__ void __launch_bounds__(256)
sno_score_kernel(const float* __restrict__ v, float* __restrict__ s) {
  const int n = blockIdx.x * 256 + threadIdx.x;
  if (n >= NPTS) return;
  float acc = 0.f;
#pragma unroll 8
  for (int h = 0; h < HD; ++h) acc += v[n * HD + h] * v[n * HD + h];
  s[n] = sqrtf(acc);
}

// ---------------------------------------------------------------------------
// Top-512 by score (single block): binary-search threshold + bitonic sort.
// ---------------------------------------------------------------------------
__global__ void __launch_bounds__(1024)
sno_topk_kernel(const float* __restrict__ s, int* __restrict__ topOut) {
  __shared__ float red[1024];
  __shared__ int scnt;
  __shared__ int nsel;
  __shared__ float selS[NSP];
  __shared__ int selI[NSP];

  const int tid = threadIdx.x;

  float m = -1.f;
  for (int i = tid; i < NPTS; i += 1024) m = fmaxf(m, s[i]);
  red[tid] = m;
  __syncthreads();
  for (int off = 512; off > 0; off >>= 1) {
    if (tid < off) red[tid] = fmaxf(red[tid], red[tid + off]);
    __syncthreads();
  }
  const float smax = red[0];
  __syncthreads();

  float lo = 0.0f, hi = smax;
  for (int it = 0; it < 32; ++it) {
    const float mid = 0.5f * (lo + hi);
    if (tid == 0) scnt = 0;
    __syncthreads();
    int c = 0;
    for (int i = tid; i < NPTS; i += 1024) c += (s[i] > mid) ? 1 : 0;
    atomicAdd(&scnt, c);
    __syncthreads();
    if (scnt > NSP) lo = mid; else hi = mid;
    __syncthreads();
  }

  if (tid == 0) nsel = 0;
  __syncthreads();
  for (int i = tid; i < NPTS; i += 1024) {
    if (s[i] > hi) {
      const int pos = atomicAdd(&nsel, 1);
      if (pos < NSP) { selS[pos] = s[i]; selI[pos] = i; }
    }
  }
  __syncthreads();
  if (nsel < NSP) {
    for (int i = tid; i < NPTS; i += 1024) {
      const float si = s[i];
      if (si > lo && si <= hi) {
        const int pos = atomicAdd(&nsel, 1);
        if (pos < NSP) { selS[pos] = si; selI[pos] = i; }
      }
    }
  }
  __syncthreads();
  const int have = (nsel < NSP) ? nsel : NSP;
  if (tid < NSP && tid >= have) { selS[tid] = -3.4e38f; selI[tid] = 0; }
  __syncthreads();

  // Bitonic sort, descending by score, ties -> ascending index
  for (int k = 2; k <= NSP; k <<= 1) {
    for (int j = k >> 1; j > 0; j >>= 1) {
      if (tid < NSP) {
        const int ixj = tid ^ j;
        if (ixj > tid) {
          const bool up = ((tid & k) == 0);
          const float si = selS[tid], sj = selS[ixj];
          const int ii = selI[tid], ij = selI[ixj];
          const bool jFirst = (sj > si) || (sj == si && ij < ii);
          if (up == jFirst) {
            selS[tid] = sj; selS[ixj] = si;
            selI[tid] = ij; selI[ixj] = ii;
          }
        }
      }
      __syncthreads();
    }
  }
  if (tid < NSP) topOut[tid] = selI[tid];
}

// ---------------------------------------------------------------------------
// Anchor 8-NN per point over 512 anchors (anchors cached in LDS).
// ---------------------------------------------------------------------------
__global__ void __launch_bounds__(256)
sno_ann_kernel(const float* __restrict__ coords, const int* __restrict__ topIdx,
               int* __restrict__ annOut) {
  __shared__ float ax[NSP], ay[NSP], az[NSP];
  const int tid = threadIdx.x;
  for (int a = tid; a < NSP; a += 256) {
    const int ai = topIdx[a];
    ax[a] = coords[ai * 3 + 0];
    ay[a] = coords[ai * 3 + 1];
    az[a] = coords[ai * 3 + 2];
  }
  __syncthreads();
  const int n = blockIdx.x * 256 + tid;
  if (n >= NPTS) return;
  const float px = coords[n * 3], py = coords[n * 3 + 1], pz = coords[n * 3 + 2];
  float bd[KANC];
  int bi[KANC];
#pragma unroll
  for (int m = 0; m < KANC; ++m) { bd[m] = 3.4e38f; bi[m] = 0; }
  for (int a = 0; a < NSP; ++a) {
    const float dx = ax[a] - px, dy = ay[a] - py, dz = az[a] - pz;
    const float d = dx * dx + dy * dy + dz * dz;
    if (d < bd[KANC - 1]) {
      float cd = d;
      int ci = a;
#pragma unroll
      for (int m = 0; m < KANC; ++m) {
        if (cd < bd[m]) {
          const float td = bd[m]; const int ti = bi[m];
          bd[m] = cd; bi[m] = ci; cd = td; ci = ti;
        }
      }
    }
  }
#pragma unroll
  for (int m = 0; m < KANC; ++m) annOut[n * KANC + m] = bi[m];
}

// ---------------------------------------------------------------------------
// local_A = softmax_j( dot(keys[n], queries[ann[n,j]]) / sqrt(H) )
// ---------------------------------------------------------------------------
__global__ void __launch_bounds__(256)
sno_logits_kernel(const float* __restrict__ keys, const float* __restrict__ q,
                  const int* __restrict__ ann, float* __restrict__ outA) {
  const int n = blockIdx.x * 256 + threadIdx.x;
  if (n >= NPTS) return;
  int aj[KANC];
#pragma unroll
  for (int j = 0; j < KANC; ++j) aj[j] = ann[n * KANC + j] * HD;
  float acc[KANC];
#pragma unroll
  for (int j = 0; j < KANC; ++j) acc[j] = 0.f;
  for (int h = 0; h < HD; ++h) {
    const float kh = keys[n * HD + h];
#pragma unroll
    for (int j = 0; j < KANC; ++j) acc[j] += kh * q[aj[j] + h];
  }
  float mx = -3.4e38f;
#pragma unroll
  for (int j = 0; j < KANC; ++j) { acc[j] *= 0.125f; mx = fmaxf(mx, acc[j]); }
  float se = 0.f;
#pragma unroll
  for (int j = 0; j < KANC; ++j) { acc[j] = expf(acc[j] - mx); se += acc[j]; }
  const float inv = 1.0f / se;
#pragma unroll
  for (int j = 0; j < KANC; ++j) outA[n * KANC + j] = acc[j] * inv;
}

// ---------------------------------------------------------------------------
// Host-side launch
// ---------------------------------------------------------------------------
extern "C" void kernel_launch(void* const* d_in, const int* in_sizes, int n_in,
                              void* d_out, int out_size, void* d_ws,
                              size_t ws_size, hipStream_t stream) {
  (void)in_sizes; (void)n_in; (void)out_size; (void)ws_size;

  const float* coords = (const float*)d_in[0];
  const float* feat = (const float*)d_in[1];
  const int* idx = (const int*)d_in[2];
  // d_in[3] = n_sp scalar (512, hardcoded)
  const float* lift_w = (const float*)d_in[4];
  const float* lift_b = (const float*)d_in[5];
  const float* g1_w = (const float*)d_in[6];
  const float* g1_b = (const float*)d_in[7];
  const float* g2_w = (const float*)d_in[8];
  const float* g2_b = (const float*)d_in[9];
  const float* g3_w = (const float*)d_in[10];
  const float* g3_b = (const float*)d_in[11];
  const float* W_w = (const float*)d_in[12];
  const float* ln_g = (const float*)d_in[13];
  const float* ln_b = (const float*)d_in[14];
  const float* key_w = (const float*)d_in[15];
  const float* key_b = (const float*)d_in[16];
  const float* q_w = (const float*)d_in[17];
  const float* q_b = (const float*)d_in[18];

  // Output layout (flat concat, ints stored as int32 bit patterns):
  // local_A[N,8] | anchor_nn[N,8] | top_idx[512] | v[N,64] | idx[N,16]
  float* out = (float*)d_out;
  float* outA = out;
  int* outAnn = (int*)(out + (size_t)NPTS * KANC);
  int* outTop = (int*)(out + (size_t)NPTS * KANC * 2);
  float* outV = out + (size_t)NPTS * KANC * 2 + NSP;
  void* outIdx = (void*)(out + (size_t)NPTS * KANC * 2 + NSP + (size_t)NPTS * HD);

  // Workspace layout (bytes)
  char* ws = (char*)d_ws;
  _Float16* pk1 = (_Float16*)(ws + 0);        // 5*4*32*16 halves = 20480 B
  _Float16* pk2 = (_Float16*)(ws + 20480);    // 8192 B
  _Float16* pkw = (_Float16*)(ws + 28672);    // 8192 B
  _Float16* pkk = (_Float16*)(ws + 36864);    // 8192 B
  _Float16* pkq = (_Float16*)(ws + 45056);    // 8192 B
  float* scores = (float*)(ws + (1u << 16));          // 400 KB
  float* queries = (float*)(ws + (1u << 19));         // 128 KB
  float* integral = (float*)(ws + (1u << 20));        // 25.6 MB
  float* keys = (float*)(ws + (size_t)28 * 1024 * 1024);  // 25.6 MB

  // Pack weights into WMMA B-fragment layout (once per launch).
  sno_pack_kernel<<<5, 128, 0, stream>>>(g1_w, pk1, 131);
  sno_pack_kernel<<<2, 128, 0, stream>>>(g2_w, pk2, 64);
  sno_pack_kernel<<<2, 128, 0, stream>>>(W_w, pkw, 64);
  sno_pack_kernel<<<2, 128, 0, stream>>>(key_w, pkk, 64);
  sno_pack_kernel<<<2, 128, 0, stream>>>(q_w, pkq, 64);

  sno_lift_kernel<<<(NPTS * HD + 255) / 256, 256, 0, stream>>>(coords, feat,
                                                               lift_w, lift_b,
                                                               outV);

  for (int t = 0; t < TLAY; ++t) {
    sno_green_kernel<<<NPTS / 4, 128, 0, stream>>>(outV, coords, idx, pk1, g1_b,
                                                   pk2, g2_b, g3_w, g3_b,
                                                   integral);
    sno_update_kernel<<<NPTS / 32, 64, 0, stream>>>(outV, integral, pkw,
                                                    ln_g + t * HD, ln_b + t * HD);
  }

  sno_score_kernel<<<(NPTS + 255) / 256, 256, 0, stream>>>(outV, scores);
  sno_topk_kernel<<<1, 1024, 0, stream>>>(scores, outTop);
  sno_ann_kernel<<<(NPTS + 255) / 256, 256, 0, stream>>>(coords, outTop, outAnn);

  sno_gemm16_kernel<<<NPTS / 32, 64, 0, stream>>>(outV, pkk, key_b, keys,
                                                  nullptr, NPTS);
  sno_gemm16_kernel<<<NSP / 32, 64, 0, stream>>>(outV, pkq, q_b, queries,
                                                 outTop, NSP);

  sno_logits_kernel<<<(NPTS + 255) / 256, 256, 0, stream>>>(keys, queries,
                                                            outAnn, outA);

  hipMemcpyAsync(outIdx, (const void*)idx, (size_t)NPTS * KNB * sizeof(int),
                 hipMemcpyDeviceToDevice, stream);
}
